// MambaLayer_56332791054492
// MI455X (gfx1250) — compile-verified, matrix operand order checked by reading, not run on hardware
//
#include <hip/hip_runtime.h>
#include <hip/hip_bf16.h>

#define NT    32768      // tokens (32*32*32)
#define DIMC  128        // model dim
#define DI    256        // d_inner
#define DST   16         // d_state
#define CH    128        // chunk length for scan
#define NCHK  (NT / CH)  // 256 chunks

#define LOG2E 1.4426950408889634f

typedef __attribute__((ext_vector_type(16))) __bf16          v16bf;
typedef __attribute__((ext_vector_type(16))) unsigned short  v16us;
typedef __attribute__((ext_vector_type(8)))  float           v8f;

// raw v_exp_f32 (base-2 exponential unit)
__device__ inline float fexp2(float x) { return __builtin_amdgcn_exp2f(x); }

// float -> bf16 bits, round-to-nearest-even
__device__ inline unsigned short f2bfu(float f) {
  union { float f; unsigned u; } v; v.f = f;
  unsigned r = v.u + 0x7FFFu + ((v.u >> 16) & 1u);
  return (unsigned short)(r >> 16);
}

// Load one 16x32 bf16 fragment (A layout per CDNA5 ISA 7.12.2; B mirrored
// with N on lanes). base points at [row0][kbase] of a row-major ushort(bf16)
// tile with leading dimension ld. Per-lane data is contiguous pairs, so the
// compiler coalesces these into ds_load_b128s.
__device__ inline v16bf load_frag(const unsigned short* base, int ld, int lane) {
  const int m = lane & 15, half = lane >> 4;
  const unsigned short* rp = base + m * ld + 8 * half;
  v16us t;
#pragma unroll
  for (int v = 0; v < 4; ++v) {
    t[2 * v]     = rp[2 * v];          // VGPR v   : K = 2v, 2v+1 (+8*half)
    t[2 * v + 1] = rp[2 * v + 1];
    t[8 + 2 * v] = rp[16 + 2 * v];     // VGPR 4+v : K = 16+2v (+8*half)
    t[9 + 2 * v] = rp[17 + 2 * v];
  }
  return __builtin_bit_cast(v16bf, t);
}

// permutation: mode 0 = identity, 1 = reverse, 2 = strided (S = 1<<sh)
__device__ inline int permf(int p, int mode, int sh) {
  if (mode == 0) return p;
  if (mode == 1) return NT - 1 - p;
  const int S = 1 << sh;
  return (p & (S - 1)) * (NT >> sh) + (p >> sh);
}

// silu via exp2 (fold log2e once)
__device__ inline float siluf(float v) { return v / (1.f + fexp2(-v * LOG2E)); }

// ---------------------------------------------------------------------------
// Kernel A: fused LayerNorm + in_proj GEMM (M=64 tokens, N=128 slab, K=128)
// grid (NT/64, 4), block 256
// ---------------------------------------------------------------------------
__global__ __launch_bounds__(256) void k_ln_inproj(
    const float* __restrict__ x, const float* __restrict__ g,
    const float* __restrict__ bia, const float* __restrict__ w,
    float* __restrict__ xi, float* __restrict__ z) {
  __shared__ __align__(16) char smem[33280 + 16384 + 2560];
  float*          xs   = (float*)smem;                       // [128][65] f32
  unsigned short* abf  = (unsigned short*)(smem + 33280);    // [64][128] bf16
  float*          psum = (float*)(smem + 33280 + 16384);     // [4][64]
  float*          psq  = psum + 256;                         // [4][64]
  float*          mu   = psq + 256;                          // [64]
  float*          rs   = mu + 64;                            // [64]
  unsigned short* wbf  = (unsigned short*)smem;              // reuse: [128][128] bf16

  const int tid = threadIdx.x;
  const int t0  = blockIdx.x * 64;
  const int nb  = blockIdx.y;          // which 128-wide output slab (0..3)

  __builtin_prefetch(&w[(size_t)nb * 128 * DIMC + tid * 64], 0, 1);

  // stage x[c][t0..t0+63] -> LDS (padded ld=65 to avoid bank conflicts)
#pragma unroll
  for (int it = 0; it < 32; ++it) {
    int idx = tid + it * 256;
    int c = idx >> 6, t = idx & 63;
    xs[c * 65 + t] = x[c * NT + t0 + t];
  }
  __syncthreads();
  {  // per-token mean/var (4 partial threads per token)
    int t = tid & 63, part = tid >> 6;
    float s = 0.f, q = 0.f;
#pragma unroll
    for (int j = 0; j < 32; ++j) {
      float v = xs[(part * 32 + j) * 65 + t];
      s += v; q += v * v;
    }
    psum[part * 64 + t] = s; psq[part * 64 + t] = q;
  }
  __syncthreads();
  if (tid < 64) {
    float s = psum[tid] + psum[64 + tid] + psum[128 + tid] + psum[192 + tid];
    float q = psq[tid] + psq[64 + tid] + psq[128 + tid] + psq[192 + tid];
    float m = s * (1.f / 128.f);
    float v = q * (1.f / 128.f) - m * m;
    mu[tid] = m;
    rs[tid] = rsqrtf(v + 1e-5f);
  }
  __syncthreads();
  // normalized bf16 A-tile [64 tok][128 k]
#pragma unroll
  for (int it = 0; it < 32; ++it) {
    int idx = tid + it * 256;
    int t = idx >> 7, c = idx & 127;
    float v = (xs[c * 65 + t] - mu[t]) * rs[t] * g[c] + bia[c];
    abf[t * 128 + c] = f2bfu(v);
  }
  __syncthreads();
  // weight slab -> bf16 (xs region is dead, reuse it)
  const int n0 = nb * 128;
#pragma unroll
  for (int it = 0; it < 64; ++it) {
    int idx = tid + it * 256;
    int n = idx >> 7, k = idx & 127;
    wbf[n * 128 + k] = f2bfu(w[(n0 + n) * DIMC + k]);
  }
  __syncthreads();

  // WMMA: 4 M-tiles x 8 N-tiles over 8 waves (4 tiles each), K = 128 (4 steps)
  const int lane = tid & 31, wave = tid >> 5;
  float* dst; int cbase;
  if (nb < 2) { dst = xi; cbase = nb * 128; }
  else        { dst = z;  cbase = (nb - 2) * 128; }
#pragma unroll
  for (int tt = 0; tt < 4; ++tt) {
    int tile = wave * 4 + tt;
    int mt = tile >> 3, nt = tile & 7;
    v8f acc = {0.f, 0.f, 0.f, 0.f, 0.f, 0.f, 0.f, 0.f};
#pragma unroll
    for (int kk = 0; kk < 4; ++kk) {
      v16bf a = load_frag(abf + (mt * 16) * 128 + kk * 32, 128, lane);
      v16bf b = load_frag(wbf + (nt * 16) * 128 + kk * 32, 128, lane);
      acc = __builtin_amdgcn_wmma_f32_16x16x32_bf16(false, a, false, b,
                                                    (short)0, acc, false, false);
    }
    int m0 = t0 + mt * 16 + 8 * (lane >> 4);
    int n  = cbase + nt * 16 + (lane & 15);
#pragma unroll
    for (int r = 0; r < 8; ++r) dst[(size_t)(m0 + r) * DI + n] = acc[r];
  }
}

// ---------------------------------------------------------------------------
// Kernel B: fused permuted-gather + causal depthwise conv(k=4) + SiLU +
//           xproj WMMA (N=40 padded to 48) + dt-proj + softplus + B/C split
// grid NT/32, block 256 (thread == channel)
// ---------------------------------------------------------------------------
__global__ __launch_bounds__(256) void k_conv_xproj(
    const float* __restrict__ xi, const float* __restrict__ cw,
    const float* __restrict__ cb, const float* __restrict__ xw,
    const float* __restrict__ dtw, const float* __restrict__ dtbv,
    float* __restrict__ xc, float* __restrict__ dtg,
    float* __restrict__ Bm, float* __restrict__ Cm,
    int dir, int mode, int sh) {
  __shared__ __align__(16) unsigned short abf[32 * 256];   // 16 KB
  __shared__ __align__(16) unsigned short wbf[48 * 256];   // 24 KB
  __shared__ __align__(16) float          dbl[32 * 48];    //  6 KB

  const int tid = threadIdx.x, lane = tid & 31, wave = tid >> 5;
  const int t0 = blockIdx.x * 32;
  const int c  = tid;

  // depthwise causal conv over the permuted sequence + SiLU
  const float w0 = cw[dir * DI * 4 + c * 4 + 0];
  const float w1 = cw[dir * DI * 4 + c * 4 + 1];
  const float w2 = cw[dir * DI * 4 + c * 4 + 2];
  const float w3 = cw[dir * DI * 4 + c * 4 + 3];
  const float cbi = cb[dir * DI + c];
  float h0 = 0.f, h1 = 0.f, h2 = 0.f;
  {
    int p = t0 - 3;
    if (p >= 0)     h0 = xi[(size_t)permf(p, mode, sh) * DI + c];
    if (p + 1 >= 0) h1 = xi[(size_t)permf(p + 1, mode, sh) * DI + c];
    if (p + 2 >= 0) h2 = xi[(size_t)permf(p + 2, mode, sh) * DI + c];
  }
  for (int t = 0; t < 32; ++t) {
    int p = t0 + t;
    float h3 = xi[(size_t)permf(p, mode, sh) * DI + c];
    float v = cbi + w0 * h0 + w1 * h1 + w2 * h2 + w3 * h3;
    v = siluf(v);
    xc[(size_t)p * DI + c] = v;
    abf[t * 256 + c] = f2bfu(v);
    h0 = h1; h1 = h2; h2 = h3;
  }
  // xproj weights: 48x256 bf16 (rows >= 40 zero-padded)
#pragma unroll
  for (int it = 0; it < 48; ++it) {
    int idx = tid + it * 256;
    int n = idx >> 8, k = idx & 255;
    float v = (n < 40) ? xw[dir * 40 * DI + n * DI + k] : 0.f;
    wbf[idx] = f2bfu(v);
  }
  __syncthreads();

  // WMMA: 2 M-tiles x 3 N-tiles, waves 0..5, K = 256 (8 steps)
  if (wave < 6) {
    int mt = wave / 3, nt = wave % 3;
    v8f acc = {0.f, 0.f, 0.f, 0.f, 0.f, 0.f, 0.f, 0.f};
#pragma unroll
    for (int kk = 0; kk < 8; ++kk) {
      v16bf a = load_frag(abf + (mt * 16) * 256 + kk * 32, 256, lane);
      v16bf b = load_frag(wbf + (nt * 16) * 256 + kk * 32, 256, lane);
      acc = __builtin_amdgcn_wmma_f32_16x16x32_bf16(false, a, false, b,
                                                    (short)0, acc, false, false);
    }
    int half = lane >> 4, n = nt * 16 + (lane & 15);
#pragma unroll
    for (int r = 0; r < 8; ++r) dbl[(mt * 16 + 8 * half + r) * 48 + n] = acc[r];
  }
  __syncthreads();

  // dt = softplus(dbl[:, :8] @ dtw^T + b); split B/C columns
  float dw[8];
#pragma unroll
  for (int r = 0; r < 8; ++r) dw[r] = dtw[dir * DI * 8 + c * 8 + r];
  const float db = dtbv[dir * DI + c];
  for (int t = 0; t < 32; ++t) {
    float s = db;
#pragma unroll
    for (int r = 0; r < 8; ++r) s += dbl[t * 48 + r] * dw[r];
    // softplus = log2(1 + 2^(s*log2e)) / log2e
    float sp = (s > 20.f) ? s
             : __log2f(1.f + fexp2(s * LOG2E)) * (1.f / LOG2E);
    dtg[(size_t)(t0 + t) * DI + c] = sp;
    if (c < 16)        Bm[(size_t)(t0 + t) * DST + c]        = dbl[t * 48 + 8 + c];
    else if (c < 32)   Cm[(size_t)(t0 + t) * DST + (c - 16)] = dbl[t * 48 + 24 + (c - 16)];
  }
}

// ---------------------------------------------------------------------------
// Kernel C1: chunk-local scan (zero init) -> per-chunk decay product + endpoint
// grid NCHK, block 256 (thread == channel; 16 states in registers).
// A2[s] pre-folds log2e so the hot loop is exactly one v_exp_f32 + 3 fma-class
// ops per state per token.
// ---------------------------------------------------------------------------
__global__ __launch_bounds__(256) void k_scan1(
    const float* __restrict__ dtg, const float* __restrict__ xc,
    const float* __restrict__ Bm, const float* __restrict__ Alog,
    float* __restrict__ Ap, float* __restrict__ He, int dir) {
  __shared__ float Bs[CH * DST];   // 8 KB
  const int c = threadIdx.x, k = blockIdx.x, p0 = k * CH;
#pragma unroll
  for (int it = 0; it < 8; ++it) {
    int idx = c + it * 256;
    Bs[idx] = Bm[(size_t)p0 * DST + idx];
  }
  float A2[DST], h[DST], ap[DST];
#pragma unroll
  for (int s = 0; s < DST; ++s) {
    A2[s] = -__expf(Alog[dir * DI * DST + c * DST + s]) * LOG2E;
    h[s] = 0.f; ap[s] = 1.f;
  }
  __syncthreads();
  for (int t = 0; t < CH; ++t) {
    float dt = dtg[(size_t)(p0 + t) * DI + c];
    float du = dt * xc[(size_t)(p0 + t) * DI + c];
#pragma unroll
    for (int s = 0; s < DST; ++s) {
      float dA = fexp2(dt * A2[s]);
      ap[s] *= dA;
      h[s] = h[s] * dA + du * Bs[t * DST + s];
    }
  }
  const size_t base = ((size_t)k * DI + c) * DST;
#pragma unroll
  for (int s = 0; s < DST; ++s) { Ap[base + s] = ap[s]; He[base + s] = h[s]; }
}

// ---------------------------------------------------------------------------
// Kernel C2: chunk-summary scan (256 steps over 4096 lanes)
// grid 16, block 256
// ---------------------------------------------------------------------------
__global__ __launch_bounds__(256) void k_scan2(
    const float* __restrict__ Ap, const float* __restrict__ He,
    float* __restrict__ Hi) {
  const int gid = blockIdx.x * 256 + threadIdx.x;  // (c,s) pair, 0..4095
  float h = 0.f;
  for (int k = 0; k < NCHK; ++k) {
    Hi[(size_t)k * 4096 + gid] = h;
    h = h * Ap[(size_t)k * 4096 + gid] + He[(size_t)k * 4096 + gid];
  }
}

// ---------------------------------------------------------------------------
// Kernel C3: chunk replay with correct initial state -> y, unpermuted+accum.
// grid NCHK, block 256
// ---------------------------------------------------------------------------
__global__ __launch_bounds__(256) void k_scan3(
    const float* __restrict__ dtg, const float* __restrict__ xc,
    const float* __restrict__ Bm, const float* __restrict__ Cm,
    const float* __restrict__ Alog, const float* __restrict__ Dp,
    const float* __restrict__ Hi, float* __restrict__ ytot,
    int dir, int mode, int sh, int accum) {
  __shared__ float Bs[CH * DST];   // 8 KB
  __shared__ float Cs[CH * DST];   // 8 KB
  const int c = threadIdx.x, k = blockIdx.x, p0 = k * CH;
#pragma unroll
  for (int it = 0; it < 8; ++it) {
    int idx = c + it * 256;
    Bs[idx] = Bm[(size_t)p0 * DST + idx];
    Cs[idx] = Cm[(size_t)p0 * DST + idx];
  }
  float A2[DST], h[DST];
#pragma unroll
  for (int s = 0; s < DST; ++s) {
    A2[s] = -__expf(Alog[dir * DI * DST + c * DST + s]) * LOG2E;
    h[s]  = Hi[(size_t)k * 4096 + c * DST + s];
  }
  const float Dv = Dp[dir * DI + c];
  __syncthreads();
  for (int t = 0; t < CH; ++t) {
    float dt = dtg[(size_t)(p0 + t) * DI + c];
    float u  = xc[(size_t)(p0 + t) * DI + c];
    float du = dt * u;
    float y = 0.f;
#pragma unroll
    for (int s = 0; s < DST; ++s) {
      float dA = fexp2(dt * A2[s]);
      h[s] = h[s] * dA + du * Bs[t * DST + s];
      y += h[s] * Cs[t * DST + s];
    }
    y += u * Dv;
    int pp = permf(p0 + t, mode, sh);           // inverse-permute on write
    float* d = &ytot[(size_t)pp * DI + c];      // contiguous 1KB row per token
    if (accum) *d += y; else *d = y;
  }
}

// ---------------------------------------------------------------------------
// Kernel E: fused y*silu(z) + out_proj WMMA + transposed write + residual
// grid (NT/32, 2), block 256
// ---------------------------------------------------------------------------
__global__ __launch_bounds__(256) void k_out(
    const float* __restrict__ ytot, const float* __restrict__ z,
    const float* __restrict__ w, const float* __restrict__ x,
    float* __restrict__ out) {
  __shared__ __align__(16) unsigned short abf[32 * 256];   // 16 KB
  __shared__ __align__(16) unsigned short wbf[64 * 256];   // 32 KB
  const int tid = threadIdx.x, lane = tid & 31, wave = tid >> 5;
  const int t0 = blockIdx.x * 32, n0 = blockIdx.y * 64;

  __builtin_prefetch(&w[(size_t)n0 * DI + tid * 64], 0, 1);

#pragma unroll
  for (int it = 0; it < 32; ++it) {
    int idx = tid + it * 256;
    int t = idx >> 8, kk = idx & 255;
    float zz = z[(size_t)(t0 + t) * DI + kk];
    float a = ytot[(size_t)(t0 + t) * DI + kk] * siluf(zz);
    abf[idx] = f2bfu(a);
  }
#pragma unroll
  for (int it = 0; it < 64; ++it) {
    int idx = tid + it * 256;
    int n = idx >> 8, kk = idx & 255;
    wbf[idx] = f2bfu(w[(size_t)(n0 + n) * DI + kk]);
  }
  __syncthreads();

  // 2 M-tiles x 4 N-tiles, one per wave, K = 256 (8 steps)
  const int mt = wave >> 2, nt = wave & 3;
  v8f acc = {0.f, 0.f, 0.f, 0.f, 0.f, 0.f, 0.f, 0.f};
#pragma unroll
  for (int kk = 0; kk < 8; ++kk) {
    v16bf a = load_frag(abf + (mt * 16) * 256 + kk * 32, 256, lane);
    v16bf b = load_frag(wbf + (nt * 16) * 256 + kk * 32, 256, lane);
    acc = __builtin_amdgcn_wmma_f32_16x16x32_bf16(false, a, false, b,
                                                  (short)0, acc, false, false);
  }
  const int half = lane >> 4;
  const int m = t0 + mt * 16 + 8 * half;       // token
  const int n = n0 + nt * 16 + (lane & 15);    // output channel
#pragma unroll
  for (int r = 0; r < 8; ++r) {
    size_t idx = (size_t)n * NT + m + r;       // out is [C][T]; +residual
    out[idx] = acc[r] + x[idx];
  }
}

// ---------------------------------------------------------------------------
extern "C" void kernel_launch(void* const* d_in, const int* in_sizes, int n_in,
                              void* d_out, int out_size, void* d_ws, size_t ws_size,
                              hipStream_t stream) {
  const float* x    = (const float*)d_in[0];
  const float* lng  = (const float*)d_in[1];
  const float* lnb  = (const float*)d_in[2];
  const float* ipw  = (const float*)d_in[3];
  const float* cw   = (const float*)d_in[4];
  const float* cb   = (const float*)d_in[5];
  const float* xw   = (const float*)d_in[6];
  const float* dtw  = (const float*)d_in[7];
  const float* dtbv = (const float*)d_in[8];
  const float* alog = (const float*)d_in[9];
  const float* Dp   = (const float*)d_in[10];
  const float* opw  = (const float*)d_in[11];
  float* out = (float*)d_out;

  float* ws = (float*)d_ws;
  size_t o = 0;
  float* xi   = ws + o; o += (size_t)NT * DI;          // 33.5 MB
  float* z    = ws + o; o += (size_t)NT * DI;
  float* ytot = ws + o; o += (size_t)NT * DI;
  float* xc   = ws + o; o += (size_t)NT * DI;          // per-dir temps
  float* dtg  = ws + o; o += (size_t)NT * DI;
  float* Bm   = ws + o; o += (size_t)NT * DST;
  float* Cm   = ws + o; o += (size_t)NT * DST;
  float* Ap   = ws + o; o += (size_t)NCHK * DI * DST;
  float* He   = ws + o; o += (size_t)NCHK * DI * DST;
  float* Hi   = ws + o; o += (size_t)NCHK * DI * DST;
  (void)ws_size; (void)in_sizes; (void)n_in; (void)out_size;

  k_ln_inproj<<<dim3(NT / 64, 4), 256, 0, stream>>>(x, lng, lnb, ipw, xi, z);

  const int modes[4] = {0, 1, 2, 2};
  const int shs[4]   = {0, 0, 5, 10};   // S = 32, S = 1024 strided scans
  for (int d = 0; d < 4; ++d) {
    k_conv_xproj<<<NT / 32, 256, 0, stream>>>(xi, cw, cb, xw, dtw, dtbv,
                                              xc, dtg, Bm, Cm, d, modes[d], shs[d]);
    k_scan1<<<NCHK, 256, 0, stream>>>(dtg, xc, Bm, alog, Ap, He, d);
    k_scan2<<<16, 256, 0, stream>>>(Ap, He, Hi);
    k_scan3<<<NCHK, 256, 0, stream>>>(dtg, xc, Bm, Cm, alog, Dp, Hi, ytot,
                                      d, modes[d], shs[d], d > 0 ? 1 : 0);
  }
  k_out<<<dim3(NT / 32, 2), 256, 0, stream>>>(ytot, z, opw, x, out);
}
